// DomainAdaptiveAttention_53755810676862
// MI455X (gfx1250) — compile-verified, matrix operand order checked by reading, not run on hardware
//
#include <hip/hip_runtime.h>

typedef __attribute__((ext_vector_type(16))) _Float16 v16h;
typedef __attribute__((ext_vector_type(8)))  _Float16 v8h;
typedef __attribute__((ext_vector_type(8)))  float    v8f;
typedef __attribute__((ext_vector_type(4)))  unsigned int u32x4;
typedef __attribute__((ext_vector_type(8)))  int i32x8;
typedef __attribute__((ext_vector_type(4)))  int i32x4;

namespace {
constexpr int  kE = 768, kB = 2, kS = 2048;
constexpr long XH_ELT = (long)kB * kS * kE;   // 3,145,728 halves (x / Q / K / V / AO)
constexpr long WP_ELT = (long)kE * kE;        // 589,824 halves per packed weight
constexpr unsigned kPanelBytes = 24u * 2048u * 2u;  // 96KB B panel in LDS
}

union F16x16 { v16h v; _Float16 e[16]; };
union F32x8  { v8f  v; float    e[8];  };

__device__ __forceinline__ v8f wmma_f16(v16h a, v16h b, v8f c) {
  return __builtin_amdgcn_wmma_f32_16x16x32_f16(false, a, false, b, (short)0, c, false, false);
}

// A-fragment loader: 16-bit A 16x32 layout. Lane holds row (lane&15); its 16
// halves are K = {0..7,16..23} (+8 for upper half-wave) -> two contiguous b128s.
__device__ __forceinline__ v16h load_a32(const _Float16* p) {
  F16x16 a;
  v8h lo = *(const v8h*)p;
  v8h hi = *(const v8h*)(p + 16);
#pragma unroll
  for (int i = 0; i < 8; ++i) { a.e[i] = lo[i]; a.e[8 + i] = hi[i]; }
  return a.v;
}

#if defined(__has_builtin)
#if __has_builtin(__builtin_amdgcn_tensor_load_to_lds) && __has_builtin(__builtin_amdgcn_s_wait_tensorcnt)
#define USE_TDM 1
#endif
#endif

#ifdef USE_TDM
// 2D TDM: copy rows x w_elts (2-byte elements, row stride in elements) to LDS.
__device__ __forceinline__ void tdm_load_2d(unsigned lds_byte_off, const void* gsrc,
                                            unsigned w_elts, unsigned rows,
                                            unsigned long row_stride_elts) {
  unsigned long ga = (unsigned long)gsrc;
  u32x4 g0;
  g0[0] = 1u;                                               // count=1, user descriptor
  g0[1] = lds_byte_off;                                     // lds_addr
  g0[2] = (unsigned)ga;                                     // global_addr[31:0]
  g0[3] = (unsigned)((ga >> 32) & 0x01FFFFFFu) | (2u << 30);// global_addr[56:32] | type=2
  i32x8 g1;
  g1[0] = 0x10000;                                          // wg_mask=0, data_size=1 (2B)
  g1[1] = (int)((w_elts & 0xFFFFu) << 16);                  // tensor_dim0[15:0]
  g1[2] = (int)((w_elts >> 16) | ((rows & 0xFFFFu) << 16)); // dim0[31:16] | dim1[15:0]
  g1[3] = (int)((rows >> 16) | ((w_elts & 0xFFFFu) << 16)); // dim1[31:16] | tile_dim0
  g1[4] = (int)(rows & 0xFFFFu);                            // tile_dim1 | tile_dim2=0
  g1[5] = (int)(row_stride_elts & 0xFFFFFFFFul);            // dim0_stride[31:0]
  g1[6] = (int)((row_stride_elts >> 32) & 0xFFFFul);        // dim0_stride[47:32]
  g1[7] = 0;
  i32x4 z4;
  z4[0] = 0; z4[1] = 0; z4[2] = 0; z4[3] = 0;
  i32x8 z8;
#pragma unroll
  for (int i = 0; i < 8; ++i) z8[i] = 0;
  __builtin_amdgcn_tensor_load_to_lds(g0, g1, z4, z4, z8, 0);
}
#endif

__global__ void cvt_f16_kernel(const float* __restrict__ src, _Float16* __restrict__ dst, long n) {
  long i = (long)blockIdx.x * blockDim.x + threadIdx.x;
  if (i < n) dst[i] = (_Float16)src[i];
}

// Pack W[768][768] (k rows, n cols) into B-fragment tiles [kt=24][nt=48][512]:
// tile element (lane,i) = W[kt*32 + kof(lane,i)][nt*16 + lane%16]
__global__ void pack_w_kernel(const float* __restrict__ W, _Float16* __restrict__ Wp) {
  int idx  = blockIdx.x * blockDim.x + threadIdx.x;    // < 589824
  int i    = idx & 15;
  int lane = (idx >> 4) & 31;
  int tile = idx >> 9;
  int nt = tile % 48;
  int kt = tile / 48;
  int k = kt * 32 + ((i < 8) ? i : i + 8) + ((lane & 16) ? 8 : 0);
  int n = nt * 16 + (lane & 15);
  Wp[idx] = (_Float16)W[k * 768 + n];
}

// Pack V [24 heads][2048][64] f16 into per-head A-fragment tiles [kt=64][dt=4][512]
__global__ void pack_v_kernel(const _Float16* __restrict__ Vb, _Float16* __restrict__ Vp) {
  int idx  = blockIdx.x * blockDim.x + threadIdx.x;    // < 3145728
  int i    = idx & 15;
  int lane = (idx >> 4) & 31;
  int tile = (idx >> 9) & 255;                         // 64 kt * 4 dt
  int head = idx >> 17;                                // 2048*64 = 2^17 per head
  int dt = tile & 3;
  int kt = tile >> 2;
  int k = kt * 32 + ((i < 8) ? i : i + 8) + ((lane & 16) ? 8 : 0);
  int d = dt * 16 + (lane & 15);
  Vp[idx] = Vb[((long)head << 17) + k * 64 + d];
}

__global__ void dscale_kernel(const float* __restrict__ de, const float* __restrict__ Wd,
                              const float* __restrict__ bd, float* __restrict__ dsc) {
  int t = threadIdx.x;
  if (t >= 24) return;
  int b = t / 12, h = t % 12;
  float s = bd[h];
  for (int i = 0; i < 64; ++i) s += de[b * 64 + i] * Wd[i * 12 + h];
  dsc[t] = s * 0.125f;   // fold in 1/sqrt(64)
}

// C[4096,768] = A(f16 row-major) @ Wpacked + bias.
// Block = 4 waves -> 128(M) x 64(N). B panel (24 x 2048 halves = 96KB) staged in
// LDS once per block via TDM; wave computes 32x64, ping-pong A prefetch.
__global__ __launch_bounds__(128) void gemm_wmma_kernel(
    const _Float16* __restrict__ A, const _Float16* __restrict__ Wp,
    const float* __restrict__ bias, _Float16* __restrict__ out_h,
    float* __restrict__ out_f) {
  extern __shared__ _Float16 smem[];                   // [24][2048] B panel
  const int lane  = threadIdx.x & 31;
  const int wave  = threadIdx.x >> 5;
  const int mbase = blockIdx.x * 128 + wave * 32;
  const int nbase = blockIdx.y * 64;
  const int lm    = lane & 15;
  const int hi8   = (lane & 16) ? 8 : 0;
  const int nt0   = nbase >> 4;

#ifdef USE_TDM
  if (wave == 0) {
    tdm_load_2d((unsigned)(size_t)(&smem[0]), Wp + (long)nt0 * 512,
                2048u, 24u, 48ul * 512ul);
    __builtin_amdgcn_s_wait_tensorcnt(0);
  }
  __syncthreads();
#else
  {  // cooperative fallback copy: 128 threads x v8h
    const _Float16* src = Wp + (long)nt0 * 512;
    for (int r = 0; r < 24; ++r) {
      for (int c = threadIdx.x * 8; c < 2048; c += 128 * 8)
        *(v8h*)(smem + r * 2048 + c) = *(const v8h*)(src + (long)r * 48 * 512 + c);
    }
  }
  __syncthreads();
#endif

  const _Float16* a0p = A + (long)(mbase + lm) * 768 + hi8;
  const _Float16* a1p = a0p + 16 * 768;

  v8f acc[2][4] = {};
  v16h aA0 = load_a32(a0p), aA1 = load_a32(a1p);     // even-kt buffer
  v16h aB0, aB1;                                      // odd-kt buffer

  auto kstep = [&](int kt, v16h& x0, v16h& x1) {
    const _Float16* bp = smem + kt * 2048 + lane * 16;
#pragma unroll
    for (int t = 0; t < 4; ++t) {
      v16h b = *(const v16h*)(bp + t * 512);          // ds_load from shared panel
      acc[0][t] = wmma_f16(x0, b, acc[0][t]);
      acc[1][t] = wmma_f16(x1, b, acc[1][t]);
    }
  };

  for (int kt = 0; kt < 24; kt += 2) {
    aB0 = load_a32(a0p + (kt + 1) * 32);              // prefetch odd kt
    aB1 = load_a32(a1p + (kt + 1) * 32);
    kstep(kt, aA0, aA1);
    if (kt + 2 < 24) {                                // prefetch next even kt
      aA0 = load_a32(a0p + (kt + 2) * 32);
      aA1 = load_a32(a1p + (kt + 2) * 32);
    }
    kstep(kt + 1, aB0, aB1);
  }

#pragma unroll
  for (int m = 0; m < 2; ++m) {
    const int mb = mbase + m * 16;
#pragma unroll
    for (int t = 0; t < 4; ++t) {
      F32x8 u; u.v = acc[m][t];
      const int cidx = nbase + t * 16 + lm;
      const float bv = bias[cidx];
#pragma unroll
      for (int j = 0; j < 8; ++j) {
        int r = mb + hi8 + j;                    // C layout: M = j + 8*(upper half)
        float val = u.e[j] + bv;
        if (out_f) out_f[(long)r * 768 + cidx] = val;
        if (out_h) {                             // scatter into [B,H,S,64]
          int bb = r >> 11, s = r & 2047;
          int h = cidx >> 6, d = cidx & 63;
          out_h[(((long)(bb * 12 + h) << 11) + s) * 64 + d] = (_Float16)val;
        }
      }
    }
  }
}

// Transposed flash attention: per wave one 16-row q tile, 64 chunks of 32 keys.
// S^T = K @ Q^T keeps the softmax (key) axis per-lane; P^T converts in-lane to
// the B-fragment for out^T = V^T @ P^T. K chunks ping-pong between two register
// buffers; V fragments issued early, consumed after softmax.
__global__ __launch_bounds__(128) void attn_kernel(
    const _Float16* __restrict__ Qb, const _Float16* __restrict__ Kb,
    const _Float16* __restrict__ Vp, const float* __restrict__ dsc,
    _Float16* __restrict__ AO) {
  const int lane = threadIdx.x & 31;
  const int wave = threadIdx.x >> 5;
  const int bh   = blockIdx.y;                 // 0..23
  const int qt   = blockIdx.x * 4 + wave;      // 0..127
  const int lm   = lane & 15;
  const int hi8  = (lane & 16) ? 8 : 0;
  const long hbase = (long)bh << 17;           // 2048*64 per head
  const _Float16* Q = Qb + hbase;
  const _Float16* K = Kb + hbase;
  const _Float16* V = Vp + hbase;
  const float sc = dsc[bh];
  const int qbase = qt * 16;

  // Q^T B-fragments for the two d-halves (loaded once, reused 64x)
  F16x16 bq0, bq1;
  {
    const _Float16* qp = Q + (qbase + lm) * 64;
#pragma unroll
    for (int i = 0; i < 16; ++i) {
      int d = ((i < 8) ? i : i + 8) + hi8;
      bq0.e[i] = qp[d];
      bq1.e[i] = qp[32 + d];
    }
  }

  const _Float16* kbase = K + (long)lm * 64 + hi8;
  auto loadK = [&](int kc, v16h* ka) {
    const _Float16* kp = kbase + (long)kc * 32 * 64;
    ka[0] = load_a32(kp);                 // tile0, d 0..31
    ka[1] = load_a32(kp + 32);            // tile0, d 32..63
    ka[2] = load_a32(kp + 1024);          // tile1, d 0..31
    ka[3] = load_a32(kp + 1024 + 32);     // tile1, d 32..63
  };

  v8f o[4] = {};
  float mrow = -1e30f, lrow = 0.f;

  // one 32-key chunk; prefetches chunk `pf` into kapf between scores & softmax
  auto chunk = [&](int kc, v16h* ka, int pf, v16h* kapf) {
    const _Float16* vp = V + (long)kc * 2048 + lane * 16;
    v16h vv0 = *(const v16h*)(vp);
    v16h vv1 = *(const v16h*)(vp + 512);
    v16h vv2 = *(const v16h*)(vp + 1024);
    v16h vv3 = *(const v16h*)(vp + 1536);

    v8f st0 = {}, st1 = {};
    st0 = wmma_f16(ka[0], bq0.v, st0);
    st0 = wmma_f16(ka[1], bq1.v, st0);
    st1 = wmma_f16(ka[2], bq0.v, st1);
    st1 = wmma_f16(ka[3], bq1.v, st1);

    if (pf < 64) loadK(pf, kapf);

    F32x8 s0, s1; s0.v = st0; s1.v = st1;
    float cmax = -1e30f;
#pragma unroll
    for (int j = 0; j < 8; ++j) {
      s0.e[j] *= sc; s1.e[j] *= sc;
      cmax = fmaxf(cmax, fmaxf(s0.e[j], s1.e[j]));
    }
    cmax = fmaxf(cmax, __shfl_xor(cmax, 16, 32));   // combine half-waves
    float mnew = fmaxf(mrow, cmax);
    float corr = __expf(mrow - mnew);
    F16x16 pb;                                       // P^T B-fragment, in-lane
    float psum = 0.f;
#pragma unroll
    for (int j = 0; j < 8; ++j) {
      float e0 = __expf(s0.e[j] - mnew);
      float e1 = __expf(s1.e[j] - mnew);
      psum += e0 + e1;
      pb.e[j]     = (_Float16)e0;                    // k' = j    + hi8
      pb.e[8 + j] = (_Float16)e1;                    // k' = j+16 + hi8
    }
    lrow = lrow * corr + psum;
    mrow = mnew;

#pragma unroll
    for (int t = 0; t < 4; ++t) {
      F32x8 uo; uo.v = o[t];
#pragma unroll
      for (int j = 0; j < 8; ++j) uo.e[j] *= corr;
      v16h av = (t == 0) ? vv0 : (t == 1) ? vv1 : (t == 2) ? vv2 : vv3;
      o[t] = wmma_f16(av, pb.v, uo.v);
    }
  };

  v16h kaA[4], kaB[4];
  loadK(0, kaA);
  for (int kc = 0; kc < 64; kc += 2) {
    chunk(kc,     kaA, kc + 1, kaB);   // even chunk, prefetch odd
    chunk(kc + 1, kaB, kc + 2, kaA);   // odd chunk, prefetch next even
  }

  float ltot = lrow + __shfl_xor(lrow, 16, 32);
  float inv = 1.f / ltot;
  const int tokrow  = (bh / 12) * 2048 + qbase + lm;
  const int colbase = (bh % 12) * 64;
#pragma unroll
  for (int t = 0; t < 4; ++t) {
    F32x8 uo; uo.v = o[t];
#pragma unroll
    for (int j = 0; j < 8; ++j) {
      int d = colbase + t * 16 + hi8 + j;
      AO[(long)tokrow * 768 + d] = (_Float16)(uo.e[j] * inv);
    }
  }
}

extern "C" void kernel_launch(void* const* d_in, const int* in_sizes, int n_in,
                              void* d_out, int out_size, void* d_ws, size_t ws_size,
                              hipStream_t stream) {
  (void)in_sizes; (void)n_in; (void)out_size; (void)ws_size;
  const float* x  = (const float*)d_in[0];
  const float* de = (const float*)d_in[1];
  const float* Wq = (const float*)d_in[2];
  const float* bq = (const float*)d_in[3];
  const float* Wk = (const float*)d_in[4];
  const float* bk = (const float*)d_in[5];
  const float* Wv = (const float*)d_in[6];
  const float* bv = (const float*)d_in[7];
  const float* Wd = (const float*)d_in[8];
  const float* bd = (const float*)d_in[9];
  const float* Wp = (const float*)d_in[10];
  const float* bp = (const float*)d_in[11];
  float* out = (float*)d_out;

  _Float16* ws16 = (_Float16*)d_ws;
  _Float16* xh  = ws16;
  _Float16* wqp = xh  + XH_ELT;
  _Float16* wkp = wqp + WP_ELT;
  _Float16* wvp = wkp + WP_ELT;
  _Float16* wpp = wvp + WP_ELT;
  _Float16* qb  = wpp + WP_ELT;
  _Float16* kb  = qb  + XH_ELT;
  _Float16* vb  = kb  + XH_ELT;
  _Float16* vpk = vb  + XH_ELT;
  _Float16* ao  = vpk + XH_ELT;
  float*    dsc = (float*)(ao + XH_ELT);

  cvt_f16_kernel<<<(XH_ELT + 255) / 256, 256, 0, stream>>>(x, xh, XH_ELT);
  pack_w_kernel<<<(WP_ELT + 255) / 256, 256, 0, stream>>>(Wq, wqp);
  pack_w_kernel<<<(WP_ELT + 255) / 256, 256, 0, stream>>>(Wk, wkp);
  pack_w_kernel<<<(WP_ELT + 255) / 256, 256, 0, stream>>>(Wv, wvp);
  pack_w_kernel<<<(WP_ELT + 255) / 256, 256, 0, stream>>>(Wp, wpp);
  dscale_kernel<<<1, 32, 0, stream>>>(de, Wd, bd, dsc);

  dim3 gg(32, 12);   // 4096/128 x 768/64
  gemm_wmma_kernel<<<gg, 128, kPanelBytes, stream>>>(xh, wqp, bq, qb, nullptr);
  gemm_wmma_kernel<<<gg, 128, kPanelBytes, stream>>>(xh, wkp, bk, kb, nullptr);
  gemm_wmma_kernel<<<gg, 128, kPanelBytes, stream>>>(xh, wvp, bv, vb, nullptr);
  pack_v_kernel<<<(XH_ELT + 255) / 256, 256, 0, stream>>>(vb, vpk);

  attn_kernel<<<dim3(32, 24), 128, 0, stream>>>(qb, kb, vpk, dsc, ao);

  gemm_wmma_kernel<<<gg, 128, kPanelBytes, stream>>>(ao, wpp, bp, nullptr, out);
}
